// ViscoelasticMaterialModel_22050362097744
// MI455X (gfx1250) — matrix operand order repaired and checked by reading, not compiled, not run on hardware
//
#include <hip/hip_runtime.h>
#include <hip/hip_bf16.h>
#include <math.h>

typedef __attribute__((ext_vector_type(16))) _Float16 v16h;
typedef __attribute__((ext_vector_type(8)))  float    v8f;

#define BB   2048
#define TT   256
#define NN   128
#define WID  32
#define HID  50
#define DTS  0.01f
#define BN32 (BB*NN*WID)          /* 8388608 floats per activation buffer */
#define PI2_OVER_N (6.28318530717958647692f/128.f)

__device__ __forceinline__ float gelu_tanh(float x) {
    float x3 = x * x * x;
    float t  = tanhf(0.7978845608028654f * (x + 0.044715f * x3));
    return 0.5f * x * (1.f + t);
}

__device__ __forceinline__ float softplusf(float x) {
    if (x > 20.f) return x;
    return log1pf(__expf(x));
}

/* ---- lift: h[b,n,o] = sum_i micro[b,i,n]*lw[i,o] + lb[o]  (h in [B,N,C]) ---- */
__global__ __launch_bounds__(256) void k_lift(const float* __restrict__ s0,
                                              const float* __restrict__ s1,
                                              const float* __restrict__ lw,
                                              const float* __restrict__ lb,
                                              float* __restrict__ h, int din) {
    int idx = blockIdx.x * 256 + threadIdx.x;   /* b*4096 + n*32 + o */
    int o = idx & 31;
    int bn = idx >> 5;                          /* b*128 + n */
    float acc = lb[o] + s0[bn] * lw[o];
    if (din == 2) acc += s1[bn] * lw[32 + o];
    h[idx] = acc;
}

/* ---- truncated DFT as WMMA:
 * xf[b,i,m] = sum_n h[b,n,i] * e^{-i 2pi m n /128}, m<4.
 * GEMM view: M = channel rows (16/tile), K = n (4 steps of 32),
 * twiddle B cols 0..3 = cos(2pi m n/N), cols 4..7 = -sin, cols 8..15 = 0.
 * One wave per (b, 16-channel) tile -> 4 chained v_wmma_f32_16x16x32_f16. ---- */
__global__ __launch_bounds__(128) void k_dft_wmma(const float* __restrict__ h,
                                                  float* __restrict__ xf_re,
                                                  float* __restrict__ xf_im) {
    const int wave  = (blockIdx.x << 2) | (threadIdx.x >> 5);  /* 0..4095 */
    const int lane  = threadIdx.x & 31;
    const int b     = wave >> 1;
    const int ibase = (wave & 1) << 4;
    const int g     = lane >> 4;
    const int l15   = lane & 15;

    const float* hcol = h + b * 4096 + ibase + l15;   /* A row = channel */
    const float mmul  = (float)((l15 < 4) ? l15 : (l15 - 4));
    const bool  is_re = (l15 < 4);
    const bool  col_on = (l15 < 8);

    v8f C = {0.f, 0.f, 0.f, 0.f, 0.f, 0.f, 0.f, 0.f};
    #pragma unroll
    for (int kb = 0; kb < 4; ++kb) {
        /* A fragment: 16x32 f16, row = l15, halfs: k = 8g..8g+7, 16+8g..16+8g+7 */
        v16h A;
        #pragma unroll
        for (int e = 0; e < 8; ++e) {
            A[e]     = (_Float16)hcol[(kb * 32 + g * 8 + e) * 32];
            A[e + 8] = (_Float16)hcol[(kb * 32 + 16 + g * 8 + e) * 32];
        }
        /* B fragment: twiddles, col = l15, element e -> k = 8g + e (e<8), 16 + 8g + (e-8) (e>=8) */
        v16h Btw;
        #pragma unroll
        for (int e = 0; e < 16; ++e) {
            int n = kb * 32 + ((e < 8) ? (g * 8 + e) : (16 + g * 8 + (e - 8)));
            float ang = mmul * (float)n * PI2_OVER_N;
            float s, c;
            __sincosf(ang, &s, &c);
            float val = is_re ? c : -s;
            Btw[e] = (_Float16)(col_on ? val : 0.f);
        }
        C = __builtin_amdgcn_wmma_f32_16x16x32_f16(false, A, false, Btw,
                                                   (short)0, C, false, false);
    }
    /* D: VGPR j -> (channel = ibase + j + 8g, col = l15); store divergence is
     * after all WMMAs so EXEC-all-ones holds during the matrix ops. */
    if (col_on) {
        int m = is_re ? l15 : (l15 - 4);
        float* dst = (is_re ? xf_re : xf_im) + b * 128 + m;
        #pragma unroll
        for (int j = 0; j < 8; ++j) {
            int i = ibase + j + 8 * g;
            dst[i * 4] = C[j];
        }
    }
}

/* ---- spectral mix: om[b,o,m] = sum_i xf[b,i,m] * (wr + j wi)[i,o,m] ---- */
__global__ __launch_bounds__(256) void k_specmul(const float* __restrict__ xf_re,
                                                 const float* __restrict__ xf_im,
                                                 const float* __restrict__ spec_r,
                                                 const float* __restrict__ spec_i,
                                                 int l,
                                                 float* __restrict__ om_re,
                                                 float* __restrict__ om_im) {
    int idx = blockIdx.x * 256 + threadIdx.x;   /* b*128 + o*4 + m */
    int b = idx >> 7, rem = idx & 127, o = rem >> 2, m = rem & 3;
    const float* xr = xf_re + b * 128;
    const float* xi = xf_im + b * 128;
    const float* wr = spec_r + l * 4096;        /* (i*32+o)*4+m */
    const float* wi = spec_i + l * 4096;
    float ar = 0.f, ai = 0.f;
    #pragma unroll 4
    for (int i = 0; i < 32; ++i) {
        float xrv = xr[i * 4 + m], xiv = xi[i * 4 + m];
        float wrv = wr[(i * 32 + o) * 4 + m], wiv = wi[(i * 32 + o) * 4 + m];
        ar += xrv * wrv - xiv * wiv;
        ai += xrv * wiv + xiv * wrv;
    }
    om_re[idx] = ar;
    om_im[idx] = ai;
}

/* ---- WMMA layer: h_out = gelu( iDFT(om) + pw_b + h @ pw_w )
 * Weight tile (4 KB) staged to LDS via global_load_async_to_lds_b128
 * (ASYNCcnt) and consumed via ds loads. iDFT + bias fused into the C
 * operand; GELU fused into the store. Two WMMAs cover o = 0..31. ---- */
__global__ __launch_bounds__(128) void k_pw_wmma(const float* __restrict__ h,
                                                 const float* __restrict__ om_re,
                                                 const float* __restrict__ om_im,
                                                 const float* __restrict__ pw_w,
                                                 const float* __restrict__ pw_b,
                                                 float* __restrict__ h_out) {
    __shared__ float wlds[1024];                 /* 32x32 f32 weight tile */
    const int tid   = threadIdx.x;
    const int wave  = (blockIdx.x << 2) | (tid >> 5);  /* 0..16383 */
    const int lane  = tid & 31;
    const int b     = wave >> 3;
    const int nbase = (wave & 7) << 4;
    const int g     = lane >> 4;
    const int l15   = lane & 15;

    /* kick off async copy: each of 128 threads moves 32 bytes (2 x B128);
     * offset:16 advances both the LDS and global addresses per the ISA. */
    {
        unsigned ldsoff = (unsigned)(size_t)(&wlds[0]) + (unsigned)tid * 32u;
        unsigned goff   = (unsigned)tid * 32u;
        asm volatile("global_load_async_to_lds_b128 %0, %1, %2\n\t"
                     "global_load_async_to_lds_b128 %0, %1, %2 offset:16"
                     :: "v"(ldsoff), "v"(goff), "s"(pw_w)
                     : "memory");
    }

    /* A fragment: 16x32 f16, row M = l15, halfs: K = 8g..8g+7, 16+8g..16+8g+7 */
    const float* hrow = h + b * 4096 + (nbase + l15) * 32;
    v16h A;
    #pragma unroll
    for (int e = 0; e < 8; ++e) {
        A[e]     = (_Float16)hrow[g * 8 + e];
        A[e + 8] = (_Float16)hrow[16 + g * 8 + e];
    }

    /* C fragments: spec (4-mode irfft) + bias. C VGPR j <-> (M=j+8g, N=l15). */
    const float* omr = om_re + b * 128;
    const float* omi = om_im + b * 128;
    int o0 = l15, o1 = l15 + 16;
    float r00 = omr[o0*4+0], r01 = omr[o0*4+1], r02 = omr[o0*4+2], r03 = omr[o0*4+3];
    float i01 = omi[o0*4+1], i02 = omi[o0*4+2], i03 = omi[o0*4+3];
    float r10 = omr[o1*4+0], r11 = omr[o1*4+1], r12 = omr[o1*4+2], r13 = omr[o1*4+3];
    float i11 = omi[o1*4+1], i12 = omi[o1*4+2], i13 = omi[o1*4+3];
    float bias0 = pw_b[o0], bias1 = pw_b[o1];

    v8f C0, C1;
    #pragma unroll
    for (int j = 0; j < 8; ++j) {
        int n = nbase + j + 8 * g;
        float th = (float)n * PI2_OVER_N;
        float c1 = __cosf(th), s1 = __sinf(th);
        float c2 = c1 * c1 - s1 * s1, s2 = 2.f * s1 * c1;
        float c3 = c2 * c1 - s2 * s1, s3 = s2 * c1 + c2 * s1;
        float sp0 = (r00 + 2.f * (r01*c1 - i01*s1 + r02*c2 - i02*s2 + r03*c3 - i03*s3)) * (1.f/128.f);
        float sp1 = (r10 + 2.f * (r11*c1 - i11*s1 + r12*c2 - i12*s2 + r13*c3 - i13*s3)) * (1.f/128.f);
        C0[j] = sp0 + bias0;
        C1[j] = sp1 + bias1;
    }

    /* complete this wave's async transfers, then block-level visibility */
    asm volatile("s_wait_asynccnt 0x0" ::: "memory");
    __syncthreads();

    /* B fragments from LDS: col = o_base + l15, element e -> K = 8g + e (e<8), 16 + 8g + (e-8) (e>=8) */
    v16h B0, B1;
    #pragma unroll
    for (int e = 0; e < 16; ++e) {
        int k = (e < 8) ? (g * 8 + e) : (16 + g * 8 + (e - 8));
        B0[e] = (_Float16)wlds[k * 32 + l15];
        B1[e] = (_Float16)wlds[k * 32 + 16 + l15];
    }

    C0 = __builtin_amdgcn_wmma_f32_16x16x32_f16(false, A, false, B0, (short)0, C0, false, false);
    C1 = __builtin_amdgcn_wmma_f32_16x16x32_f16(false, A, false, B1, (short)0, C1, false, false);

    float* orow = h_out + b * 4096;
    #pragma unroll
    for (int j = 0; j < 8; ++j) {
        int n = nbase + j + 8 * g;
        orow[n * 32 + o0] = gelu_tanh(C0[j]);
        orow[n * 32 + o1] = gelu_tanh(C1[j]);
    }
}

/* ---- head: mean over n, 32->64 gelu, 64->1 ---- */
__global__ __launch_bounds__(64) void k_head(const float* __restrict__ h,
                                             const float* __restrict__ f1w,
                                             const float* __restrict__ f1b,
                                             const float* __restrict__ f2w,
                                             const float* __restrict__ f2b,
                                             float* __restrict__ fout) {
    __shared__ float pooled[32];
    __shared__ float y[64];
    int b = blockIdx.x, t = threadIdx.x;
    if (t < 32) {
        const float* src = h + b * 4096 + t;
        float s = 0.f;
        #pragma unroll 4
        for (int n = 0; n < 128; ++n) s += src[n * 32];
        pooled[t] = s * (1.f / 128.f);
    }
    __syncthreads();
    {
        float acc = f1b[t];
        #pragma unroll 4
        for (int c = 0; c < 32; ++c) acc += pooled[c] * f1w[c * 64 + t];
        y[t] = gelu_tanh(acc);
    }
    __syncthreads();
    if (t == 0) {
        float acc = f2b[0];
        #pragma unroll 4
        for (int k = 0; k < 64; ++k) acc += y[k] * f2w[k];
        fout[b] = acc;
    }
}

/* ---- sequential scan: one wave32 per batch row, lanes split HID=50,
 * analytic grads of energy MLP and both PICNNs, butterfly reductions. ---- */
__global__ __launch_bounds__(256) void k_timeloop(
        const float* __restrict__ e, const float* __restrict__ ed,
        const float* __restrict__ fv,
        const float* __restrict__ W1, const float* __restrict__ b1,
        const float* __restrict__ W2,
        const float* __restrict__ Wy0_1, const float* __restrict__ Wu0_1,
        const float* __restrict__ b0_1,  const float* __restrict__ Wz1_1,
        const float* __restrict__ Wy1_1,
        const float* __restrict__ Wy0_2, const float* __restrict__ Wu0_2,
        const float* __restrict__ b0_2,  const float* __restrict__ Wz1_2,
        const float* __restrict__ Wy1_2,
        float* __restrict__ out) {
    int b    = (blockIdx.x << 3) | (threadIdx.x >> 5);   /* wave id == batch row */
    int lane = threadIdx.x & 31;

    float f1 = fv[b], f2 = fv[BB + b], f3 = fv[2 * BB + b], f4 = fv[3 * BB + b];

    float ceu[2], cev[2], ez[2], qu[2], qv[2];
    float wy1a[2], g1[2], c1[2], wy2a[2], g2[2], c2[2];
    #pragma unroll
    for (int s = 0; s < 2; ++s) {
        int j = lane + 32 * s;
        bool ok = (j < HID);
        int jc = ok ? j : 0;
        float w0  = W1[jc], w1v = W1[HID + jc];
        float w2v = W1[2 * HID + jc], w3v = W1[3 * HID + jc];
        float ww2 = W2[jc];
        ceu[s]  = ok ? w0 : 0.f;
        cev[s]  = ok ? w1v : 0.f;
        ez[s]   = ok ? (f1 * w2v + f3 * w3v + b1[jc]) : 0.f;
        qu[s]   = ok ? 2.f * w0 * ww2 : 0.f;
        qv[s]   = ok ? 2.f * w1v * ww2 : 0.f;
        wy1a[s] = ok ? Wy0_1[jc] : 0.f;
        g1[s]   = ok ? (f2 * Wu0_1[jc] + b0_1[jc]) : 0.f;
        c1[s]   = ok ? 2.f * Wy0_1[jc] * softplusf(Wz1_1[jc]) : 0.f;
        wy2a[s] = ok ? Wy0_2[jc] : 0.f;
        g2[s]   = ok ? (f4 * Wu0_2[jc] + b0_2[jc]) : 0.f;
        c2[s]   = ok ? 2.f * Wy0_2[jc] * softplusf(Wz1_2[jc]) : 0.f;
    }
    float wy11 = Wy1_1[0], wy12 = Wy1_2[0];

    float v = 0.f;   /* xi */
    const float* eb  = e  + b * TT;
    const float* edb = ed + b * TT;
    float* so = out + b * TT;
    float* xo = out + BB * TT + b * TT;

    for (int t = 0; t < TT; ++t) {
        float u = eb[t], p = edb[t];
        float S1 = 0.f, S2 = 0.f, S3 = 0.f;
        #pragma unroll
        for (int s = 0; s < 2; ++s) {
            float z = u * ceu[s] + v * cev[s] + ez[s];
            float r = fmaxf(z, 0.f);
            S1 += r * qu[s];
            S2 += r * qv[s];
            float a = p * wy1a[s] + g1[s];
            S3 += fmaxf(a, 0.f) * c1[s];
        }
        #pragma unroll
        for (int m = 16; m >= 1; m >>= 1) {
            S1 += __shfl_xor(S1, m, 32);
            S2 += __shfl_xor(S2, m, 32);
            S3 += __shfl_xor(S3, m, 32);
        }
        /* s_eq = S1 ; d = S2 ; s_neq = -(S3 + Wy1_1) ; stress = s_eq - s_neq */
        float stress = S1 + S3 + wy11;
        float qq = -S2;
        float S4 = 0.f;
        #pragma unroll
        for (int s = 0; s < 2; ++s) {
            float a = qq * wy2a[s] + g2[s];
            S4 += fmaxf(a, 0.f) * c2[s];
        }
        #pragma unroll
        for (int m = 16; m >= 1; m >>= 1) S4 += __shfl_xor(S4, m, 32);
        if (lane == 0) { so[t] = stress; xo[t] = v; }
        v += DTS * (S4 + wy12);
    }
}

extern "C" void kernel_launch(void* const* d_in, const int* in_sizes, int n_in,
                              void* d_out, int out_size, void* d_ws, size_t ws_size,
                              hipStream_t stream) {
    (void)in_sizes; (void)n_in; (void)out_size; (void)ws_size;
#define P(i) ((const float*)d_in[(i)])
    /* flat input order: e, e_dot, E, nu, then params leaves in construction order */
    const float* e   = P(0);
    const float* ed  = P(1);
    const float* E   = P(2);
    const float* nu  = P(3);

    float* ws  = (float*)d_ws;
    float* hA  = ws;
    float* hB  = hA + BN32;
    float* xfr = hB + BN32;                 /* B*32*4 */
    float* xfi = xfr + BB * 128;
    float* omr = xfi + BB * 128;
    float* omi = omr + BB * 128;
    float* fv  = omi + BB * 128;            /* 4*B scalar features */

    /* fnm1 @24, fnm2 @34, fnm3 @44 (d_in=1, field=nu), fnm4 @54 */
    for (int i = 0; i < 4; ++i) {
        int pb = 24 + 10 * i;
        const float* s0 = (i == 2) ? nu : E;
        const float* s1 = (i == 2) ? nullptr : nu;
        int din = (i == 2) ? 1 : 2;
        const float* lw  = P(pb + 0); const float* lb  = P(pb + 1);
        const float* sr  = P(pb + 2); const float* si  = P(pb + 3);
        const float* pw  = P(pb + 4); const float* pwb = P(pb + 5);
        const float* f1w = P(pb + 6); const float* f1b = P(pb + 7);
        const float* f2w = P(pb + 8); const float* f2b = P(pb + 9);
        float* fout = fv + i * BB;

        k_lift<<<BN32 / 256, 256, 0, stream>>>(s0, s1, lw, lb, hA, din);
        float* cur = hA; float* nxt = hB;
        for (int l = 0; l < 3; ++l) {
            k_dft_wmma<<<(BB * 2) / 4, 128, 0, stream>>>(cur, xfr, xfi);
            k_specmul<<<(BB * 128) / 256, 256, 0, stream>>>(xfr, xfi, sr, si, l, omr, omi);
            k_pw_wmma<<<(BB * (NN / 16)) / 4, 128, 0, stream>>>(cur, omr, omi,
                                                                pw + l * 1024, pwb + l * 32, nxt);
            float* tmp = cur; cur = nxt; nxt = tmp;
        }
        k_head<<<BB, 64, 0, stream>>>(cur, f1w, f1b, f2w, f2b, fout);
    }

    /* energy @4..7 ; picnn1 @8..15 ; picnn2 @16..23 */
    k_timeloop<<<BB / 8, 256, 0, stream>>>(e, ed, fv,
        P(4), P(5), P(6),
        P(10), P(11), P(12), P(13), P(14),
        P(18), P(19), P(20), P(21), P(22),
        (float*)d_out);
#undef P
}
